// TranxParser_86303072846411
// MI455X (gfx1250) — matrix operand order repaired and checked by reading, not compiled, not run on hardware
//
#include <hip/hip_runtime.h>
#include <hip/hip_bf16.h>

// ---------------------------------------------------------------------------
// TranX parser decoder step, MI455X (gfx1250, wave32, WMMA)
// B=1024 L=100 D=256 DIN=704 V=32000 R=129
// ---------------------------------------------------------------------------

#define B_SZ   1024
#define L_SZ   100
#define D_SZ   256
#define DIN_SZ 704
#define V_SZ   32000
#define R_SZ   129
#define KCAT   (DIN_SZ + D_SZ)   // 960, LSTM fused K
#define R_PAD  144               // 129 padded to multiple of 16

typedef __attribute__((ext_vector_type(16))) __bf16 v16bf;
typedef __attribute__((ext_vector_type(8)))  float  v8f;
typedef __attribute__((ext_vector_type(4)))  unsigned int v4u;
typedef __attribute__((ext_vector_type(8)))  int v8i_t;
typedef __attribute__((ext_vector_type(4)))  int v4i_t;

struct alignas(16) Q16 { unsigned int w[4]; };
union AFrag { v16bf v; Q16 q[2]; };

__device__ __forceinline__ __bf16 f2bf(float f) {
  union { float f; unsigned u; } in; in.f = f;
  unsigned r = (in.u + 0x7fffu + ((in.u >> 16) & 1u)) >> 16;   // RNE
  union { unsigned short s; __bf16 b; } out; out.s = (unsigned short)r;
  return out.b;
}
__device__ __forceinline__ float sigf(float x) { return 1.0f / (1.0f + __expf(-x)); }

__device__ __forceinline__ float wave_reduce_sum(float v) {
#pragma unroll
  for (int m = 16; m > 0; m >>= 1) v += __shfl_xor(v, m, 32);
  return v;
}

// ---------------------------------------------------------------------------
// Generic bf16 GEMM: C[M,N] = A[M,K] * B[N,K]^T, f32 accumulate.
// One wave = one 16x16 C tile; 8 waves / block; M = gridDim.x*16.
// Fragment layout per CDNA5 ISA 7.12.2 (16-bit A 16x32 / B 32x16 / C f32).
// ---------------------------------------------------------------------------
__global__ void gemm_bf16_tn(const __bf16* __restrict__ A,
                             const __bf16* __restrict__ B,
                             float* __restrict__ C,
                             int N, int K, int lda, int ldb, int ldc) {
  const int lane  = threadIdx.x & 31;
  const int wave  = threadIdx.x >> 5;
  const int mTile = blockIdx.x;
  const int nTile = blockIdx.y * 8 + wave;
  if (nTile * 16 >= N) return;
  const int half = lane >> 4;
  const int lr   = lane & 15;
  const __bf16* pA = A + (size_t)(mTile * 16 + lr) * lda + half * 8;
  const __bf16* pB = B + (size_t)(nTile * 16 + lr) * ldb + half * 8;
  v8f acc = {};
  for (int k0 = 0; k0 < K; k0 += 32) {
    AFrag a, b;
    a.q[0] = *(const Q16*)(pA + k0);
    a.q[1] = *(const Q16*)(pA + k0 + 16);
    b.q[0] = *(const Q16*)(pB + k0);
    b.q[1] = *(const Q16*)(pB + k0 + 16);
    acc = __builtin_amdgcn_wmma_f32_16x16x32_bf16(false, a.v, false, b.v,
                                                  (short)0, acc, false, false);
  }
  float* pC = C + (size_t)(mTile * 16 + half * 8) * ldc + (size_t)nTile * 16 + lr;
#pragma unroll
  for (int r = 0; r < 8; ++r) pC[(size_t)r * ldc] = acc[r];
}

// ---------------------------------------------------------------------------
// Register-resident-A GEMM: K = KTILES*32 known at compile time; the A panel
// (KTILES fragments) is loaded once per wave and reused across NT N-tiles.
// Halves vmem issue per wmma for the dominant [1024 x 32000 x 128] GEMM.
// ---------------------------------------------------------------------------
template<int KTILES, int NT>
__global__ void gemm_bf16_tn_areuse(const __bf16* __restrict__ A,
                                    const __bf16* __restrict__ B,
                                    float* __restrict__ C,
                                    int N, int lda, int ldb, int ldc) {
  const int lane   = threadIdx.x & 31;
  const int wave   = threadIdx.x >> 5;
  const int mTile  = blockIdx.x;
  const int nTile0 = (blockIdx.y * 8 + wave) * NT;
  if (nTile0 * 16 >= N) return;
  const int half = lane >> 4;
  const int lr   = lane & 15;

  const __bf16* pA = A + (size_t)(mTile * 16 + lr) * lda + half * 8;
  AFrag a[KTILES];
#pragma unroll
  for (int k = 0; k < KTILES; ++k) {
    a[k].q[0] = *(const Q16*)(pA + k * 32);
    a[k].q[1] = *(const Q16*)(pA + k * 32 + 16);
  }
#pragma unroll
  for (int t = 0; t < NT; ++t) {
    const int nTile = nTile0 + t;
    if (nTile * 16 < N) {
      const __bf16* pB = B + (size_t)(nTile * 16 + lr) * ldb + half * 8;
      v8f acc = {};
#pragma unroll
      for (int k = 0; k < KTILES; ++k) {
        AFrag b;
        b.q[0] = *(const Q16*)(pB + k * 32);
        b.q[1] = *(const Q16*)(pB + k * 32 + 16);
        acc = __builtin_amdgcn_wmma_f32_16x16x32_bf16(false, a[k].v, false, b.v,
                                                      (short)0, acc, false, false);
      }
      float* pC = C + (size_t)(mTile * 16 + half * 8) * ldc + (size_t)nTile * 16 + lr;
#pragma unroll
      for (int r = 0; r < 8; ++r) pC[(size_t)r * ldc] = acc[r];
    }
  }
}

// ---------------------------------------------------------------------------
// f32 -> bf16 pack, optional transpose and zero row padding.
// ---------------------------------------------------------------------------
__global__ void pack_bf16(const float* __restrict__ src, int srcld,
                          __bf16* __restrict__ dst, int dstld,
                          int outRows, int outCols, int validRows, int transpose) {
  long long total  = (long long)outRows * outCols;
  long long stride = (long long)gridDim.x * blockDim.x;
  for (long long i = (long long)blockIdx.x * blockDim.x + threadIdx.x; i < total; i += stride) {
    int r = (int)(i / outCols);
    int c = (int)(i % outCols);
    float v = 0.0f;
    if (r < validRows)
      v = transpose ? src[(size_t)c * srcld + r] : src[(size_t)r * srcld + c];
    dst[(size_t)r * dstld + c] = f2bf(v);
  }
}

__global__ void cvt_bf16(const float* __restrict__ src, __bf16* __restrict__ dst, int n) {
  int i = blockIdx.x * blockDim.x + threadIdx.x;
  if (i < n) dst[i] = f2bf(src[i]);
}

// ---------------------------------------------------------------------------
// LSTM pointwise
// ---------------------------------------------------------------------------
__global__ void lstm_point(const float* __restrict__ gates,
                           const float* __restrict__ b_ih,
                           const float* __restrict__ b_hh,
                           const float* __restrict__ c0,
                           float* __restrict__ h_out,
                           float* __restrict__ c_out,
                           __bf16* __restrict__ hc_bf) {
  int i = blockIdx.x * blockDim.x + threadIdx.x;   // over B*D
  if (i >= B_SZ * D_SZ) return;
  int b = i >> 8, d = i & 255;
  const float* g = gates + (size_t)b * (4 * D_SZ);
  float ig = sigf (g[d]             + b_ih[d]             + b_hh[d]);
  float fg = sigf (g[D_SZ + d]      + b_ih[D_SZ + d]      + b_hh[D_SZ + d]);
  float gg = tanhf(g[2 * D_SZ + d]  + b_ih[2 * D_SZ + d]  + b_hh[2 * D_SZ + d]);
  float og = sigf (g[3 * D_SZ + d]  + b_ih[3 * D_SZ + d]  + b_hh[3 * D_SZ + d]);
  float c = fg * c0[i] + ig * gg;
  float h = og * tanhf(c);
  c_out[i] = c;
  h_out[i] = h;
  hc_bf[(size_t)b * 512 + d] = f2bf(h);
}

// ---------------------------------------------------------------------------
// Tensor Data Mover: 2D f32 tile (rows x cols, row stride == cols) -> LDS.
// ---------------------------------------------------------------------------
__device__ __forceinline__ void tdm_load_2d_f32(const float* gsrc, const float* lds_dst,
                                                unsigned rows, unsigned cols) {
  unsigned long long ga = (unsigned long long)gsrc;
  unsigned la = (unsigned)(unsigned long long)lds_dst;  // flat LDS: low 32 = LDS offset
  v4u g0;
  g0[0] = 1u;                                                    // count=1
  g0[1] = la;                                                    // lds_addr
  g0[2] = (unsigned)(ga & 0xffffffffull);                        // global_addr lo
  g0[3] = (unsigned)((ga >> 32) & 0x01ffffffull) | (2u << 30);   // addr[56:32] | type=2
  v8i_t g1;
  g1[0] = (int)(2u << 16);                                       // data_size=2 (4B)
  g1[1] = (int)((cols & 0xffffu) << 16);                         // tensor_dim0 lo16
  g1[2] = (int)(((cols >> 16) & 0xffffu) | ((rows & 0xffffu) << 16));
  g1[3] = (int)(((rows >> 16) & 0xffffu) | ((cols & 0xffffu) << 16)); // tile_dim0
  g1[4] = (int)(rows & 0xffffu);                                 // tile_dim1
  g1[5] = (int)cols;                                             // dim0_stride lo32
  g1[6] = 0;
  g1[7] = 0;
  v4i_t gz; gz[0] = 0; gz[1] = 0; gz[2] = 0; gz[3] = 0;
#if __clang_major__ >= 23
  v8i_t z8;
#pragma unroll
  for (int i = 0; i < 8; ++i) z8[i] = 0;
  __builtin_amdgcn_tensor_load_to_lds(g0, g1, gz, gz, z8, 0);
#else
  __builtin_amdgcn_tensor_load_to_lds(g0, g1, gz, gz, 0);
#endif
}

// ---------------------------------------------------------------------------
// Attention: one block (8 waves) per batch row. src_enc[b] staged via TDM in
// two 50x256 f32 chunks. Row dots parallelized wave-wide with shfl reduction.
// ---------------------------------------------------------------------------
__global__ void attention_kernel(const float* __restrict__ t,
                                 const float* __restrict__ src_enc,
                                 __bf16* __restrict__ hc_bf) {
  __shared__ float stile[50 * D_SZ];   // 51200 B
  __shared__ float st[D_SZ];
  __shared__ float sc[128];
  __shared__ float salpha[128];
  const int b    = blockIdx.x;
  const int tid  = threadIdx.x;
  const int lane = tid & 31;
  const int wave = tid >> 5;
  st[tid] = t[(size_t)b * D_SZ + tid];
  const float* gsrc = src_enc + (size_t)b * L_SZ * D_SZ;

  for (int ch = 0; ch < 2; ++ch) {                 // score pass
    __syncthreads();
    if (tid < 32) {
      tdm_load_2d_f32(gsrc + (size_t)ch * 50 * D_SZ, stile, 50u, (unsigned)D_SZ);
      __builtin_amdgcn_s_wait_tensorcnt(0);
    }
    __syncthreads();
    for (int l = wave; l < 50; l += 8) {
      const float* row = stile + l * D_SZ;
      float s = 0.0f;
#pragma unroll
      for (int d = lane; d < D_SZ; d += 32) s += st[d] * row[d];
      s = wave_reduce_sum(s);
      if (lane == 0) sc[ch * 50 + l] = s;
    }
  }
  __syncthreads();
  if (tid == 0) {                                  // softmax over L=100
    float mx = -1e30f;
    for (int l = 0; l < L_SZ; ++l) mx = fmaxf(mx, sc[l]);
    float sum = 0.0f;
    for (int l = 0; l < L_SZ; ++l) { float e = __expf(sc[l] - mx); salpha[l] = e; sum += e; }
    float inv = 1.0f / sum;
    for (int l = 0; l < L_SZ; ++l) salpha[l] *= inv;
  }
  float ctx = 0.0f;
  for (int ch = 0; ch < 2; ++ch) {                 // ctx pass
    __syncthreads();
    if (tid < 32) {
      tdm_load_2d_f32(gsrc + (size_t)ch * 50 * D_SZ, stile, 50u, (unsigned)D_SZ);
      __builtin_amdgcn_s_wait_tensorcnt(0);
    }
    __syncthreads();
    for (int l = 0; l < 50; ++l) ctx += salpha[ch * 50 + l] * stile[l * D_SZ + tid];
  }
  hc_bf[(size_t)b * 512 + D_SZ + tid] = f2bf(ctx);
}

__global__ void att_tanh(const float* __restrict__ attpre,
                         float* __restrict__ att_out,
                         __bf16* __restrict__ att_bf) {
  int i = blockIdx.x * blockDim.x + threadIdx.x;
  if (i >= B_SZ * D_SZ) return;
  float a = tanhf(attpre[i]);
  att_out[i] = a;
  att_bf[i] = f2bf(a);
}

// ---------------------------------------------------------------------------
// Pointer softmax: ptr_scores[b,l] = tptr[b]·src_enc[b,l]  (src_enc L2-hot)
// ---------------------------------------------------------------------------
__global__ void ptr_softmax_kernel(const float* __restrict__ tptr,
                                   const float* __restrict__ src_enc,
                                   float* __restrict__ copy_p) {
  __shared__ float st[D_SZ];
  __shared__ float sc[128];
  const int b    = blockIdx.x;
  const int tid  = threadIdx.x;
  const int lane = tid & 31;
  const int wave = tid >> 5;
  st[tid] = tptr[(size_t)b * D_SZ + tid];
  __syncthreads();
  for (int l = wave; l < L_SZ; l += 8) {
    const float* row = src_enc + ((size_t)b * L_SZ + l) * D_SZ;
    float s = 0.0f;
#pragma unroll
    for (int d = lane; d < D_SZ; d += 32) s += st[d] * row[d];
    s = wave_reduce_sum(s);
    if (lane == 0) sc[l] = s;
  }
  __syncthreads();
  if (tid == 0) {
    float mx = -1e30f;
    for (int l = 0; l < L_SZ; ++l) mx = fmaxf(mx, sc[l]);
    float sum = 0.0f;
    for (int l = 0; l < L_SZ; ++l) { float e = __expf(sc[l] - mx); sc[l] = e; sum += e; }
    float inv = 1.0f / sum;
    for (int l = 0; l < L_SZ; ++l) copy_p[(size_t)b * L_SZ + l] = sc[l] * inv;
  }
}

__global__ void rule_logsoftmax(const float* __restrict__ logits,   // ld = R_PAD
                                float* __restrict__ out) {
  __shared__ float red[256];
  const int b = blockIdx.x, tid = threadIdx.x;
  const float* row = logits + (size_t)b * R_PAD;
  float v = (tid < R_SZ) ? row[tid] : -1e30f;
  red[tid] = v; __syncthreads();
  for (int s = 128; s > 0; s >>= 1) { if (tid < s) red[tid] = fmaxf(red[tid], red[tid + s]); __syncthreads(); }
  float mx = red[0]; __syncthreads();
  red[tid] = (tid < R_SZ) ? __expf(v - mx) : 0.0f; __syncthreads();
  for (int s = 128; s > 0; s >>= 1) { if (tid < s) red[tid] += red[tid + s]; __syncthreads(); }
  float lse = __logf(red[0]) + mx;
  if (tid < R_SZ) out[(size_t)b * R_SZ + tid] = v - lse;
}

// ---------------------------------------------------------------------------
// In-place: logits -> gate0*softmax + deterministic serial scatter of copy mass
// ---------------------------------------------------------------------------
__global__ void finalize_prim(float* __restrict__ pprim,
                              const float* __restrict__ att,
                              const float* __restrict__ w_gen,
                              const float* __restrict__ b_gen,
                              const float* __restrict__ copy_p,
                              const int* __restrict__ ids) {
  __shared__ float red[256];
  __shared__ float sg[2];
  const int b = blockIdx.x, tid = threadIdx.x;
  float* row = pprim + (size_t)b * V_SZ;
  if (tid == 0) {
    const float* a = att + (size_t)b * D_SZ;
    float s0 = b_gen[0], s1 = b_gen[1];
    for (int d = 0; d < D_SZ; ++d) { s0 += a[d] * w_gen[d]; s1 += a[d] * w_gen[D_SZ + d]; }
    float m = fmaxf(s0, s1);
    float e0 = __expf(s0 - m), e1 = __expf(s1 - m);
    float inv = 1.0f / (e0 + e1);
    sg[0] = e0 * inv; sg[1] = e1 * inv;
  }
  float mx = -1e30f;
  for (int i = tid; i < V_SZ; i += 256) mx = fmaxf(mx, row[i]);
  red[tid] = mx; __syncthreads();
  for (int s = 128; s > 0; s >>= 1) { if (tid < s) red[tid] = fmaxf(red[tid], red[tid + s]); __syncthreads(); }
  mx = red[0]; __syncthreads();
  float sum = 0.0f;
  for (int i = tid; i < V_SZ; i += 256) sum += __expf(row[i] - mx);
  red[tid] = sum; __syncthreads();
  for (int s = 128; s > 0; s >>= 1) { if (tid < s) red[tid] += red[tid + s]; __syncthreads(); }
  float scale = sg[0] / red[0];
  for (int i = tid; i < V_SZ; i += 256) row[i] = __expf(row[i] - mx) * scale;
  __syncthreads();
  if (tid == 0) {                 // serial scatter -> deterministic
    float g1 = sg[1];
    for (int l = 0; l < L_SZ; ++l)
      row[ids[(size_t)b * L_SZ + l]] += g1 * copy_p[(size_t)b * L_SZ + l];
  }
}

// ---------------------------------------------------------------------------
extern "C" void kernel_launch(void* const* d_in, const int* in_sizes, int n_in,
                              void* d_out, int out_size, void* d_ws, size_t ws_size,
                              hipStream_t stream) {
  (void)in_sizes; (void)n_in; (void)out_size; (void)ws_size;
  const float* x           = (const float*)d_in[0];
  const float* h0          = (const float*)d_in[1];
  const float* c0          = (const float*)d_in[2];
  const float* src_enc     = (const float*)d_in[3];
  const int*   ids         = (const int*)  d_in[4];
  const float* W_ih        = (const float*)d_in[5];
  const float* b_ih        = (const float*)d_in[6];
  const float* W_hh        = (const float*)d_in[7];
  const float* b_hh        = (const float*)d_in[8];
  const float* W_attn      = (const float*)d_in[9];
  const float* W_att_vec   = (const float*)d_in[10];
  const float* W_ptr       = (const float*)d_in[11];
  const float* rule_emb    = (const float*)d_in[12];
  const float* W_rule_proj = (const float*)d_in[13];
  const float* prim_emb    = (const float*)d_in[14];
  const float* W_prim_proj = (const float*)d_in[15];
  const float* w_gen       = (const float*)d_in[16];
  const float* b_gen       = (const float*)d_in[17];

  // ---- workspace carve (~25 MB) ----
  char* p = (char*)d_ws;
  auto take = [&](size_t bytes) -> char* {
    char* r = p; p += (bytes + 255) & ~(size_t)255; return r;
  };
  __bf16* Abuf    = (__bf16*)take((size_t)B_SZ * KCAT * 2);      // [x | h0]
  __bf16* Wbuf    = (__bf16*)take((size_t)(4*D_SZ) * KCAT * 2);  // [W_ih | W_hh]
  float*  gates   = (float*) take((size_t)B_SZ * 4*D_SZ * 4);
  __bf16* hc_bf   = (__bf16*)take((size_t)B_SZ * 512 * 2);       // [h | ctx]
  __bf16* WattnT  = (__bf16*)take((size_t)D_SZ * D_SZ * 2);
  float*  t_f32   = (float*) take((size_t)B_SZ * D_SZ * 4);
  __bf16* Wav     = (__bf16*)take((size_t)D_SZ * 512 * 2);
  float*  attpre  = (float*) take((size_t)B_SZ * D_SZ * 4);
  __bf16* att_bf  = (__bf16*)take((size_t)B_SZ * D_SZ * 2);
  __bf16* WrpT    = (__bf16*)take((size_t)128 * D_SZ * 2);
  __bf16* rulePad = (__bf16*)take((size_t)R_PAD * 128 * 2);
  __bf16* WppT    = (__bf16*)take((size_t)128 * D_SZ * 2);
  __bf16* prim_bf = (__bf16*)take((size_t)V_SZ * 128 * 2);
  __bf16* WptrT   = (__bf16*)take((size_t)D_SZ * D_SZ * 2);
  float*  zr_f32  = (float*) take((size_t)B_SZ * 128 * 4);
  float*  zp_f32  = (float*) take((size_t)B_SZ * 128 * 4);
  __bf16* zr_bf   = (__bf16*)take((size_t)B_SZ * 128 * 2);
  __bf16* zp_bf   = (__bf16*)take((size_t)B_SZ * 128 * 2);
  float*  tptr    = (float*) take((size_t)B_SZ * D_SZ * 4);
  float*  lrule   = (float*) take((size_t)B_SZ * R_PAD * 4);
  float*  copy_p  = (float*) take((size_t)B_SZ * L_SZ * 4);

  // ---- output layout (flat, return order) ----
  float* out       = (float*)d_out;
  float* out_rule  = out;                                   // [B,129]
  float* out_prim  = out_rule + (size_t)B_SZ * R_SZ;        // [B,32000]
  float* out_h     = out_prim + (size_t)B_SZ * V_SZ;        // [B,256]
  float* out_c     = out_h    + (size_t)B_SZ * D_SZ;        // [B,256]
  float* out_att   = out_c    + (size_t)B_SZ * D_SZ;        // [B,256]

  auto packL = [&](const float* s, int sld, __bf16* d, int dld,
                   int oR, int oC, int vR, int tr) {
    long long tot = (long long)oR * oC;
    int blocks = (int)((tot + 255) / 256);
    if (blocks > 16384) blocks = 16384;
    pack_bf16<<<dim3(blocks), dim3(256), 0, stream>>>(s, sld, d, dld, oR, oC, vR, tr);
  };

  // ---- operand packing (f32 -> bf16, transposes, padding) ----
  packL(x,           DIN_SZ, Abuf,           KCAT, B_SZ,   DIN_SZ, B_SZ,   0);
  packL(h0,          D_SZ,   Abuf + DIN_SZ,  KCAT, B_SZ,   D_SZ,   B_SZ,   0);
  packL(W_ih,        DIN_SZ, Wbuf,           KCAT, 4*D_SZ, DIN_SZ, 4*D_SZ, 0);
  packL(W_hh,        D_SZ,   Wbuf + DIN_SZ,  KCAT, 4*D_SZ, D_SZ,   4*D_SZ, 0);
  packL(W_attn,      D_SZ,   WattnT,         D_SZ, D_SZ,   D_SZ,   D_SZ,   1);
  packL(W_att_vec,   512,    Wav,            512,  D_SZ,   512,    D_SZ,   0);
  packL(W_rule_proj, 128,    WrpT,           D_SZ, 128,    D_SZ,   128,    1);
  packL(rule_emb,    128,    rulePad,        128,  R_PAD,  128,    R_SZ,   0);
  packL(W_prim_proj, 128,    WppT,           D_SZ, 128,    D_SZ,   128,    1);
  packL(prim_emb,    128,    prim_bf,        128,  V_SZ,   128,    V_SZ,   0);
  packL(W_ptr,       D_SZ,   WptrT,          D_SZ, D_SZ,   D_SZ,   D_SZ,   1);

  // ---- 1) LSTM gates: [1024 x 1024 x 960] ----
  gemm_bf16_tn<<<dim3(B_SZ/16, 8), 256, 0, stream>>>(
      Abuf, Wbuf, gates, 4*D_SZ, KCAT, KCAT, KCAT, 4*D_SZ);
  lstm_point<<<dim3(B_SZ), 256, 0, stream>>>(gates, b_ih, b_hh, c0,
                                             out_h, out_c, hc_bf);

  // ---- 2) t = h @ W_attn : [1024 x 256 x 256] ----
  gemm_bf16_tn_areuse<8, 2><<<dim3(B_SZ/16, 1), 256, 0, stream>>>(
      hc_bf, WattnT, t_f32, D_SZ, 512, D_SZ, D_SZ);

  // ---- 3) attention (TDM-staged src_enc), ctx -> hc_bf[:,256:] ----
  attention_kernel<<<dim3(B_SZ), 256, 0, stream>>>(t_f32, src_enc, hc_bf);

  // ---- 4) att = tanh([h|ctx] @ W_att_vec^T) : [1024 x 256 x 512] ----
  gemm_bf16_tn<<<dim3(B_SZ/16, 2), 256, 0, stream>>>(
      hc_bf, Wav, attpre, D_SZ, 512, 512, 512, D_SZ);
  att_tanh<<<dim3(B_SZ), 256, 0, stream>>>(attpre, out_att, att_bf);

  // ---- 5) small projections: zr, zp, tptr ----
  gemm_bf16_tn_areuse<8, 1><<<dim3(B_SZ/16, 1), 256, 0, stream>>>(
      att_bf, WrpT, zr_f32, 128, D_SZ, D_SZ, 128);
  gemm_bf16_tn_areuse<8, 1><<<dim3(B_SZ/16, 1), 256, 0, stream>>>(
      att_bf, WppT, zp_f32, 128, D_SZ, D_SZ, 128);
  gemm_bf16_tn_areuse<8, 2><<<dim3(B_SZ/16, 1), 256, 0, stream>>>(
      att_bf, WptrT, tptr, D_SZ, D_SZ, D_SZ, D_SZ);
  cvt_bf16<<<dim3((B_SZ*128 + 255)/256), 256, 0, stream>>>(zr_f32, zr_bf, B_SZ*128);
  cvt_bf16<<<dim3((B_SZ*128 + 255)/256), 256, 0, stream>>>(zp_f32, zp_bf, B_SZ*128);

  // ---- 6) rule logits: zr @ rule_emb^T : [1024 x 144 x 128] ----
  gemm_bf16_tn_areuse<4, 1><<<dim3(B_SZ/16, 2), 256, 0, stream>>>(
      zr_bf, rulePad, lrule, R_PAD, 128, 128, R_PAD);
  rule_logsoftmax<<<dim3(B_SZ), 256, 0, stream>>>(lrule, out_rule);

  // ---- 7) prim logits: zp @ prim_emb^T : [1024 x 32000 x 128] -> d_out ----
  // 2000 N-tiles, NT=4 -> 500 wave tasks per M row, 63 blocks in y.
  gemm_bf16_tn_areuse<4, 4><<<dim3(B_SZ/16, 63), 256, 0, stream>>>(
      zp_bf, prim_bf, out_prim, V_SZ, 128, 128, V_SZ);

  // ---- 8) pointer copy distribution + final mix ----
  ptr_softmax_kernel<<<dim3(B_SZ), 256, 0, stream>>>(tptr, src_enc, copy_p);
  finalize_prim<<<dim3(B_SZ), 256, 0, stream>>>(out_prim, out_att, w_gen, b_gen,
                                                copy_p, ids);
}
